// SPS1D_86071144611848
// MI455X (gfx1250) — matrix operand order repaired and checked by reading, not compiled, not run on hardware
//
#include <hip/hip_runtime.h>

typedef __attribute__((ext_vector_type(16))) _Float16 v16h;
typedef __attribute__((ext_vector_type(8)))  _Float16 v8h;
typedef __attribute__((ext_vector_type(8)))  float    v8f;
typedef __attribute__((ext_vector_type(4)))  unsigned int uint4v;
typedef __attribute__((ext_vector_type(8)))  int      int8v;
typedef __attribute__((ext_vector_type(4)))  int      int4v;

#define B_  64
#define C1_ 8
#define H_  32
#define E_  64
#define T_  16384

// ---------------------------------------------------------------------------
// conv1 (k=3, same) + BN1 : x[B][8][T] -> a1[B][T][32]  (time-major interior)
// GEMM: M=32 out-ch (2 WMMA M-tiles), N=16 times, K=24 (tap*8+ch, padded to 32)
// staging stays manual: needs f32->f16 convert + ch-major->t-major transpose
// ---------------------------------------------------------------------------
__global__ __launch_bounds__(32) void k_conv1bn(
    const float* __restrict__ x, const float* __restrict__ w1,
    const float* __restrict__ b1, const float* __restrict__ g1,
    const float* __restrict__ be1, const float* __restrict__ m1,
    const float* __restrict__ v1, float* __restrict__ a1)
{
  constexpr int TILE = 256;
  const int nblk_t = T_ / TILE;
  const int b    = blockIdx.x / nblk_t;
  const int t0   = (blockIdx.x % nblk_t) * TILE;
  const int lane = threadIdx.x;

  __shared__ __align__(16) _Float16 xT[(TILE + 2) * C1_]; // [t+halo][8ch], 16B rows
  __shared__ float2 scsh[H_];

  if (lane < H_) {
    float sc = g1[lane] * rsqrtf(v1[lane] + 1e-5f);
    scsh[lane] = make_float2(sc, be1[lane] + (b1[lane] - m1[lane]) * sc);
  }

  const float* xb = x + (size_t)b * C1_ * T_;
  for (int j = lane; j < (TILE + 2) * C1_; j += 32) {
    int i  = j / (TILE + 2);
    int tt = j - i * (TILE + 2);
    int gt = t0 - 1 + tt;
    float val = (gt >= 0 && gt < T_) ? xb[(size_t)i * T_ + gt] : 0.0f;
    xT[tt * C1_ + i] = (_Float16)val;
  }

  // A (weights) per ISA 16-bit A layout. K = k*8 + i, K=24..31 zero pad.
  const int  mrow = lane & 15;
  const bool hi   = lane >= 16;
  v16h A[2];
#pragma unroll
  for (int mt = 0; mt < 2; ++mt) {
    int o = mt * 16 + mrow;
#pragma unroll
    for (int e = 0; e < 8; ++e) {
      A[mt][e]     = (_Float16)w1[(o * C1_ + e) * 3 + (hi ? 1 : 0)];
      A[mt][8 + e] = hi ? (_Float16)0.0f : (_Float16)w1[(o * C1_ + e) * 3 + 2];
    }
  }
  __syncthreads();

  const int n = lane & 15;
  for (int nt = 0; nt < TILE / 16; ++nt) {
    const int tb = nt * 16;
    v16h Bm;
    if (!hi) { // K 0..7 : tap0 @ t-1 (row tb+n); K 16..23 : tap2 @ t+1 (row tb+n+2)
      v8h p0 = *(const v8h*)&xT[(tb + n)     * C1_];
      v8h p1 = *(const v8h*)&xT[(tb + n + 2) * C1_];
#pragma unroll
      for (int e = 0; e < 8; ++e) { Bm[e] = p0[e]; Bm[8 + e] = p1[e]; }
    } else {   // K 8..15 : tap1 @ t (row tb+n+1); K 24..31 zero pad
      v8h p2 = *(const v8h*)&xT[(tb + n + 1) * C1_];
#pragma unroll
      for (int e = 0; e < 8; ++e) { Bm[e] = p2[e]; Bm[8 + e] = (_Float16)0.0f; }
    }
    v8f d0 = {}, d1 = {};
    d0 = __builtin_amdgcn_wmma_f32_16x16x32_f16(false, A[0], false, Bm, (short)0, d0, false, false);
    d1 = __builtin_amdgcn_wmma_f32_16x16x32_f16(false, A[1], false, Bm, (short)0, d1, false, false);

    const int t  = t0 + tb + n;
    const int cb = hi ? 8 : 0;
    float* orow = a1 + ((size_t)b * T_ + t) * H_;
#pragma unroll
    for (int half = 0; half < 2; ++half) {
#pragma unroll
      for (int g4 = 0; g4 < 2; ++g4) {
        float r4[4];
#pragma unroll
        for (int e = 0; e < 4; ++e) {
          int g = g4 * 4 + e;
          float2 p = scsh[half * 16 + cb + g];
          r4[e] = (half == 0 ? d0[g] : d1[g]) * p.x + p.y;
        }
        *(float4*)&orow[half * 16 + cb + g4 * 4] = make_float4(r4[0], r4[1], r4[2], r4[3]);
      }
    }
  }
}

// ---------------------------------------------------------------------------
// LIF1: a1[B][T][32] f32 -> spikes h[B][T][32] f16 (exact 0/1 in f16)
// chunked scan: 64-step warm-up; tau=2 => state error decays 2^-64 (< f32 ulp)
// serial latency-bound stream -> explicit prefetch 64 steps (8KB) ahead
// ---------------------------------------------------------------------------
__global__ __launch_bounds__(32) void k_lif1(const float* __restrict__ a1,
                                             _Float16* __restrict__ h)
{
  constexpr int L = 512, W = 64;
  const int nchunk = T_ / L;
  const int b  = blockIdx.x / nchunk;
  const int c0 = (blockIdx.x % nchunk) * L;
  const int c  = threadIdx.x; // channel
  const float* in  = a1 + (size_t)b * T_ * H_ + c;
  _Float16*    out = h  + (size_t)b * T_ * H_ + c;
  float v = 0.0f;
  int start = c0 - W; if (start < 0) start = 0;
  for (int t = start; t < c0; ++t) {
    v = 0.5f * (v + in[(size_t)t * H_]);
    if (v >= 1.0f) v = 0.0f;
  }
  for (int t = c0; t < c0 + L; ++t) {
    __builtin_prefetch(in + (size_t)(t + 64) * H_, 0, 3); // global_prefetch_b8
    v = 0.5f * (v + in[(size_t)t * H_]);
    float s = (v >= 1.0f) ? 1.0f : 0.0f;
    v = (v >= 1.0f) ? 0.0f : v;
    out[(size_t)t * H_] = (_Float16)s;
  }
}

// ---------------------------------------------------------------------------
// conv2 (k=3, same) + BN2 : h[B][T][32] f16 -> a2[B][T][64] f32
// GEMM: M=64 (4 M-tiles), N=16 times, K=96 = 3 WMMA k-steps (K = tap*32 + ch)
// staging via Tensor Data Mover: 2D tile (32 halves x nrows), LDS row padding
// done by the TDM (pad 4 DWORDs every 16 DWORDs => 40-half rows)
// ---------------------------------------------------------------------------
__global__ __launch_bounds__(32) void k_conv2bn(
    const _Float16* __restrict__ h, const float* __restrict__ w2,
    const float* __restrict__ b2, const float* __restrict__ g2,
    const float* __restrict__ be2, const float* __restrict__ m2,
    const float* __restrict__ v2, float* __restrict__ a2)
{
  constexpr int TILE = 128;
  constexpr int RS   = 40;  // padded LDS row stride in halves (32 data + 8 pad)
  const int nblk_t = T_ / TILE;
  const int b    = blockIdx.x / nblk_t;
  const int t0   = (blockIdx.x % nblk_t) * TILE;
  const int lane = threadIdx.x;

  __shared__ __align__(16) _Float16 hT[(TILE + 2) * RS];
  __shared__ float2 scsh[E_];

  // ---- TDM descriptor: rows are 64B contiguous in global ([B][T][32] f16) ----
  int gstart   = t0 - 1;          // LDS row r <-> global time gstart + (r - lds_row0)
  int lds_row0 = 0;
  if (gstart < 0) { gstart = 0; lds_row0 = 1; }   // bottom halo: zero row 0 manually
  const int      nrows = (TILE + 2) - lds_row0;
  const uint64_t gaddr = (uint64_t)(uintptr_t)(h + ((size_t)b * T_ + gstart) * H_);
  const uint32_t laddr = (uint32_t)(uintptr_t)(&hT[lds_row0 * RS]);
  const uint32_t tdim1 = (uint32_t)(T_ - gstart); // rows in-tensor; beyond -> zero-fill

  uint4v g0;
  g0[0] = 1u;                                     // count=1 (valid), user mode
  g0[1] = laddr;                                  // lds_addr (bytes)
  g0[2] = (uint32_t)gaddr;                        // global_addr[31:0]
  g0[3] = (uint32_t)(gaddr >> 32) | (2u << 30);   // global_addr[56:32] | type=2
  int8v g1v;
  g1v[0] = (int)((1u << 16)                       // data_size = 2 bytes
               | (1u << 20)                       // pad_enable
               | (3u << 22)                       // pad_interval: every 16 DWORDs
               | (3u << 25));                     // pad_amount: 4 DWORDs
  g1v[1] = (int)(32u << 16);                      // tensor_dim0[15:0] = 32
  g1v[2] = (int)((tdim1 & 0xFFFFu) << 16);        // tensor_dim1[15:0]
  g1v[3] = (int)((tdim1 >> 16) | (32u << 16));    // tensor_dim1[31:16] | tile_dim0=32
  g1v[4] = nrows;                                 // tile_dim1
  g1v[5] = 32;                                    // tensor_dim0_stride[31:0] = 32 elems
  g1v[6] = 0;
  g1v[7] = 0;
  int4v g2v = {}; int4v g3v = {};                 // 2D: groups 2/3 unused (zeros)
  int8v g4v = {};                                 // unused VADDR4 slot (NULL per ISA)
  __builtin_amdgcn_tensor_load_to_lds(g0, g1v, g2v, g3v, g4v, 0);

  // overlap with TDM: BN constants, bottom-halo zero row, weight A-tiles
  for (int c = lane; c < E_; c += 32) {
    float sc = g2[c] * rsqrtf(v2[c] + 1e-5f);
    scsh[c] = make_float2(sc, be2[c] + (b2[c] - m2[c]) * sc);
  }
  if (lds_row0 == 1 && lane < 16)
    *(uint32_t*)&hT[lane * 2] = 0u;               // zero row 0 (t = -1)

  const int  mrow = lane & 15;
  const bool hi   = lane >= 16;
  v16h A[4][3];
#pragma unroll
  for (int mt = 0; mt < 4; ++mt) {
    int o = mt * 16 + mrow;
#pragma unroll
    for (int s = 0; s < 3; ++s) {
#pragma unroll
      for (int e = 0; e < 8; ++e) {
        int i0 = hi ? (8 + e)  : e;               // slice-local K: elems 0..7
        int i1 = hi ? (24 + e) : (16 + e);        // slice-local K: elems 8..15
        A[mt][s][e]     = (_Float16)w2[((size_t)o * H_ + i0) * 3 + s];
        A[mt][s][8 + e] = (_Float16)w2[((size_t)o * H_ + i1) * 3 + s];
      }
    }
  }
  __builtin_amdgcn_s_wait_tensorcnt(0);           // TDM tile resident in LDS
  __syncthreads();

  const int n = lane & 15;
  for (int nt = 0; nt < TILE / 16; ++nt) {
    const int tb = nt * 16;
    v8f d[4] = {};
#pragma unroll
    for (int s = 0; s < 3; ++s) {                 // tap s -> input time t + s - 1
      const int r   = tb + n + s;                 // LDS row 0 == global t0-1
      const int off = hi ? 8 : 0;                 // lanes 16-31 carry K(ch) 8..15, 24..31
      v8h p0 = *(const v8h*)&hT[r * RS + off];
      v8h p1 = *(const v8h*)&hT[r * RS + off + 16];
      v16h Bm;
#pragma unroll
      for (int e = 0; e < 8; ++e) { Bm[e] = p0[e]; Bm[8 + e] = p1[e]; }
#pragma unroll
      for (int mt = 0; mt < 4; ++mt)
        d[mt] = __builtin_amdgcn_wmma_f32_16x16x32_f16(false, A[mt][s], false, Bm,
                                                       (short)0, d[mt], false, false);
    }
    const int t  = t0 + tb + n;
    const int cb = hi ? 8 : 0;
    float* orow = a2 + ((size_t)b * T_ + t) * E_;
#pragma unroll
    for (int mt = 0; mt < 4; ++mt) {
#pragma unroll
      for (int g4 = 0; g4 < 2; ++g4) {
        float r4[4];
#pragma unroll
        for (int e = 0; e < 4; ++e) {
          int g = g4 * 4 + e;
          float2 p = scsh[mt * 16 + cb + g];
          r4[e] = d[mt][g] * p.x + p.y;
        }
        *(float4*)&orow[mt * 16 + cb + g4 * 4] = make_float4(r4[0], r4[1], r4[2], r4[3]);
      }
    }
  }
}

// ---------------------------------------------------------------------------
// LIF2: a2[B][T][64] f32 -> out[B][64][T] f32 (reference layout)
// ---------------------------------------------------------------------------
__global__ __launch_bounds__(64) void k_lif2(const float* __restrict__ a2,
                                             float* __restrict__ out)
{
  constexpr int L = 512, W = 64;
  const int nchunk = T_ / L;
  const int b  = blockIdx.x / nchunk;
  const int c0 = (blockIdx.x % nchunk) * L;
  const int c  = threadIdx.x; // 0..63
  const float* in = a2  + (size_t)b * T_ * E_ + c;
  float*       op = out + ((size_t)b * E_ + c) * T_;
  float v = 0.0f;
  int start = c0 - W; if (start < 0) start = 0;
  for (int t = start; t < c0; ++t) {
    v = 0.5f * (v + in[(size_t)t * E_]);
    if (v >= 1.0f) v = 0.0f;
  }
  float buf[4];
  for (int t = c0; t < c0 + L; ++t) {
    __builtin_prefetch(in + (size_t)(t + 64) * E_, 0, 3); // global_prefetch_b8
    v = 0.5f * (v + in[(size_t)t * E_]);
    float s = (v >= 1.0f) ? 1.0f : 0.0f;
    v = (v >= 1.0f) ? 0.0f : v;
    buf[t & 3] = s;
    if ((t & 3) == 3)
      *(float4*)&op[t - 3] = make_float4(buf[0], buf[1], buf[2], buf[3]);
  }
}

// ---------------------------------------------------------------------------
extern "C" void kernel_launch(void* const* d_in, const int* in_sizes, int n_in,
                              void* d_out, int out_size, void* d_ws, size_t ws_size,
                              hipStream_t stream)
{
  (void)in_sizes; (void)n_in; (void)out_size; (void)ws_size;
  const float* x   = (const float*)d_in[0];
  const float* w1  = (const float*)d_in[1];
  const float* b1  = (const float*)d_in[2];
  const float* g1  = (const float*)d_in[3];
  const float* be1 = (const float*)d_in[4];
  const float* m1  = (const float*)d_in[5];
  const float* v1  = (const float*)d_in[6];
  const float* w2  = (const float*)d_in[7];
  const float* b2  = (const float*)d_in[8];
  const float* g2  = (const float*)d_in[9];
  const float* be2 = (const float*)d_in[10];
  const float* m2  = (const float*)d_in[11];
  const float* v2  = (const float*)d_in[12];

  char* ws = (char*)d_ws;
  const size_t hBytes  = (size_t)B_ * H_ * T_ * sizeof(_Float16); //  64 MB
  const size_t a1Bytes = (size_t)B_ * H_ * T_ * sizeof(float);    // 128 MB
  _Float16* hbuf = (_Float16*)ws;
  float*    a1   = (float*)(ws + hBytes);
  float*    a2   = (float*)(ws + hBytes + a1Bytes);               // 256 MB

  k_conv1bn<<<dim3(B_ * (T_ / 256)), dim3(32), 0, stream>>>(x, w1, b1, g1, be1, m1, v1, a1);
  k_lif1   <<<dim3(B_ * (T_ / 512)), dim3(32), 0, stream>>>(a1, hbuf);
  k_conv2bn<<<dim3(B_ * (T_ / 128)), dim3(32), 0, stream>>>(hbuf, w2, b2, g2, be2, m2, v2, a2);
  k_lif2   <<<dim3(B_ * (T_ / 512)), dim3(64), 0, stream>>>(a2, (float*)d_out);
}
